// Seq2seq_13159779795387
// MI455X (gfx1250) — matrix operand order repaired and checked by reading, not compiled, not run on hardware
//
#include <hip/hip_runtime.h>

// ---------------------------------------------------------------------------
// 4-layer LSTM seq2seq (B=64, T=256, H=512, L=4, FUT=64) on gfx1250 (MI455X).
// Persistent kernel, wave32, v_wmma_f32_16x16x32_bf16. 128 waves in 32 blocks,
// one (batch-tile, hidden-col-tile) strip per wave covering all 4 gates.
// Device-wide software barrier (4 per timestep). c-state lives in VGPRs.
// Readout fused into the layer-3 pass via f32 global atomics.
// Branch-free activations (v_exp_f32 + v_rcp_f32); next-layer weight prefetch
// issued before each barrier so L2 latency hides under the spin.
// ---------------------------------------------------------------------------

typedef __attribute__((ext_vector_type(16))) __bf16       v16bf;
typedef __attribute__((ext_vector_type(8)))  float        v8f;
typedef __attribute__((ext_vector_type(4)))  unsigned int v4u;

constexpr int kB     = 64;          // batch
constexpr int kT     = 256;         // input timesteps
constexpr int kH     = 512;         // hidden
constexpr int kL     = 4;           // layers
constexpr int kFut   = 64;          // future steps
constexpr int kSteps = kT + kFut;   // 320
constexpr int kG     = 4 * kH;      // 2048 gate rows
constexpr int kNBlk  = 32;          // persistent blocks (>= 1 WGP each)
constexpr int kNThr  = 128;         // 4 waves/block -> 128 waves total

union Frag16 { v4u u[2]; v16bf v; };

// B operand: B[k][n] = W[n][k], W row-major [2048,512] bf16.
// lanes 0-15: N=lane, K=k0..k0+15 ; lanes 16-31: N=lane-16, K=k0+16..k0+31
__device__ __forceinline__ v16bf load_b_frag(const __bf16* W, int nrow, int k0, int lhalf) {
  const v4u* p = (const v4u*)(W + nrow * kH + k0 + lhalf * 16);
  Frag16 f; f.u[0] = p[0]; f.u[1] = p[1];
  return f.v;
}

// A operand: h row-major [64,512] bf16.
// lanes 0-15: M=lane, K=k0..k0+15
// lanes16-31: M=lane-16, K=k0+16..k0+31
__device__ __forceinline__ v16bf load_a_frag(const __bf16* A, int mrow, int k0, int lhalf) {
  Frag16 f;
  f.u[0] = *(const v4u*)(A + mrow * kH + k0 + lhalf * 16);
  f.u[1] = *(const v4u*)(A + mrow * kH + k0 + lhalf * 16 + 8);
  return f.v;
}

__device__ __forceinline__ v8f wmma_bf16(v16bf a, v16bf b, v8f c) {
  return __builtin_amdgcn_wmma_f32_16x16x32_bf16(false, a, false, b, (short)0, c,
                                                 false, false);
}

// Branch-free fast activations: v_exp_f32 + v_rcp_f32, correct +-inf limits.
__device__ __forceinline__ float sigf(float x) {
  return __builtin_amdgcn_rcpf(1.f + __expf(-x));
}
__device__ __forceinline__ float tanh_fast(float x) {
  return 1.f - 2.f * __builtin_amdgcn_rcpf(1.f + __expf(x + x));
}

// Prefetch one 32-byte B-fragment line for the next layer (no counter impact).
__device__ __forceinline__ void prefetch_b(const __bf16* W, int nrow, int k0, int lhalf) {
  __builtin_prefetch((const void*)(W + nrow * kH + k0 + lhalf * 16), 0, 3);
}

// Monotonic-counter device-wide barrier (all kNBlk blocks resident).
__device__ __forceinline__ void grid_sync(unsigned* bar, unsigned& epoch) {
  __syncthreads();
  epoch += 1;
  if (threadIdx.x == 0) {
    __threadfence();
    __hip_atomic_fetch_add(bar, 1u, __ATOMIC_RELEASE, __HIP_MEMORY_SCOPE_AGENT);
    const unsigned target = epoch * (unsigned)kNBlk;
    while (__hip_atomic_load(bar, __ATOMIC_ACQUIRE, __HIP_MEMORY_SCOPE_AGENT) < target) {
      __builtin_amdgcn_s_sleep(1);
    }
  }
  __syncthreads();
  __threadfence();
}

// One LSTM layer pass for this wave's (m-tile, 16-col) strip, all four gates.
template <bool HAS_IN, bool READOUT>
__device__ __forceinline__ void layer_pass(
    const __bf16* __restrict__ h_rec, const __bf16* __restrict__ h_in,
    __bf16* __restrict__ h_out,
    const __bf16* __restrict__ Wh, const __bf16* __restrict__ Wi,
    const float* bias4, const float* xb8, const float* wi0v4,
    float* creg /* [8] */, float wlinv, float* outAccCur,
    int mrow, int hcol0, int lrow, int lhalf, int hcol, int m0)
{
  const v8f vzero = {0.f, 0.f, 0.f, 0.f, 0.f, 0.f, 0.f, 0.f};
  v8f acc[4] = {vzero, vzero, vzero, vzero};

  #pragma unroll 2
  for (int k0 = 0; k0 < kH; k0 += 32) {
    const v16bf aR = load_a_frag(h_rec, mrow, k0, lhalf);
    v16bf aI;
    if (HAS_IN) aI = load_a_frag(h_in, mrow, k0, lhalf);
    // Issue all B-fragment loads first so they clause & overlap the WMMAs.
    v16bf bR[4], bI[4];
    #pragma unroll
    for (int g = 0; g < 4; ++g) {
      const int nrow = g * kH + hcol0 + lrow;
      bR[g] = load_b_frag(Wh, nrow, k0, lhalf);
      if (HAS_IN) bI[g] = load_b_frag(Wi, nrow, k0, lhalf);
    }
    #pragma unroll
    for (int g = 0; g < 4; ++g) {
      acc[g] = wmma_bf16(aR, bR[g], acc[g]);
      if (HAS_IN) acc[g] = wmma_bf16(aI, bI[g], acc[g]);
    }
  }

  // Fused bias + (layer-0 rank-1 input) + LSTM elementwise; c stays in VGPRs.
  // C/D layout: VGPR r -> M = m0 + lhalf*8 + r, N = hcol.
  float part[8];
  #pragma unroll
  for (int r = 0; r < 8; ++r) {
    const int b = m0 + lhalf * 8 + r;
    float gi = acc[0][r] + bias4[0];
    float gf = acc[1][r] + bias4[1];
    float gg = acc[2][r] + bias4[2];
    float go = acc[3][r] + bias4[3];
    if (!HAS_IN) {
      gi += xb8[r] * wi0v4[0];
      gf += xb8[r] * wi0v4[1];
      gg += xb8[r] * wi0v4[2];
      go += xb8[r] * wi0v4[3];
    }
    const float c2 = sigf(gf) * creg[r] + sigf(gi) * tanh_fast(gg);
    const float h2 = sigf(go) * tanh_fast(c2);
    creg[r] = c2;
    h_out[b * kH + hcol] = (__bf16)h2;
    if (READOUT) part[r] = h2 * wlinv;
  }

  if (READOUT) {
    // Reduce over this wave's 16 columns (stay inside each 16-lane half).
    #pragma unroll
    for (int r = 0; r < 8; ++r) {
      #pragma unroll
      for (int off = 8; off > 0; off >>= 1)
        part[r] += __shfl_xor(part[r], off, 32);
    }
    if (lrow == 0) {
      #pragma unroll
      for (int r = 0; r < 8; ++r)
        atomicAdd(&outAccCur[m0 + lhalf * 8 + r], part[r]);
    }
  }
}

// Warm next layer's first B tiles into the WGP cache during the barrier wait.
__device__ __forceinline__ void prefetch_layer(const __bf16* Wh, const __bf16* Wi,
                                               bool hasIn, int hcol0, int lrow,
                                               int lhalf) {
  #pragma unroll
  for (int g = 0; g < 4; ++g) {
    const int nrow = g * kH + hcol0 + lrow;
    prefetch_b(Wh, nrow, 0, lhalf);
    prefetch_b(Wh, nrow, 32, lhalf);
    if (hasIn) {
      prefetch_b(Wi, nrow, 0, lhalf);
      prefetch_b(Wi, nrow, 32, lhalf);
    }
  }
}

// fp32 -> bf16 weight repack.
__global__ void cvt_weights_kernel(const float* __restrict__ whh,
                                   const float* __restrict__ wih,
                                   __bf16* __restrict__ whhB,
                                   __bf16* __restrict__ wihB) {
  const size_t n1 = (size_t)kL * kG * kH;        // 4,194,304
  const size_t n2 = (size_t)(kL - 1) * kG * kH;  // 3,145,728
  size_t i = (size_t)blockIdx.x * blockDim.x + threadIdx.x;
  if (i < n1)           whhB[i]      = (__bf16)whh[i];
  else if (i < n1 + n2) wihB[i - n1] = (__bf16)wih[i - n1];
}

__global__ __launch_bounds__(kNThr, 1)
void lstm_persistent_kernel(const float* __restrict__ iput,   // [64,256]
                            const float* __restrict__ Wi0,    // [2048]
                            const float* __restrict__ bih,    // [4,2048]
                            const float* __restrict__ bhh,    // [4,2048]
                            const float* __restrict__ Wlin,   // [512]
                            const float* __restrict__ blin,   // [1]
                            const __bf16* __restrict__ WhhB,  // [4,2048,512]
                            const __bf16* __restrict__ WihB,  // [3,2048,512]
                            __bf16* hbuf,      // [2][L][64][512] double-buffered
                            float*  outAcc,    // [2][64] readout accumulators
                            unsigned* bar,
                            float*  out)       // [64, 320]
{
  const int tid   = threadIdx.x;
  const int wv    = tid >> 5;
  const int lane  = tid & 31;
  const int lrow  = lane & 15;
  const int lhalf = lane >> 4;
  const int wid   = blockIdx.x * (kNThr / 32) + wv;  // 0..127
  const int m0    = (wid >> 5) << 4;                 // 0/16/32/48
  const int hcol0 = (wid & 31) << 4;
  const int hcol  = hcol0 + lrow;
  const int mrow  = m0 + lrow;
  unsigned epoch = 0;

  // Hoisted per-lane constants.
  float bias[kL][4];
  #pragma unroll
  for (int j = 0; j < kL; ++j)
    #pragma unroll
    for (int g = 0; g < 4; ++g) {
      const int n = g * kH + hcol;
      bias[j][g] = bih[j * kG + n] + bhh[j * kG + n];
    }
  float wi0v[4];
  #pragma unroll
  for (int g = 0; g < 4; ++g) wi0v[g] = Wi0[g * kH + hcol];
  const float wlinv = Wlin[hcol];
  const float blin0 = blin[0];

  float creg[kL][8];   // cell state, owned by this lane forever
  #pragma unroll
  for (int j = 0; j < kL; ++j)
    #pragma unroll
    for (int r = 0; r < 8; ++r) creg[j][r] = 0.f;

  const size_t LBH = (size_t)kB * kH;

  for (int t = 0; t < kSteps; ++t) {
    const int p = t & 1, np = p ^ 1, pb = (t + 1) & 1;  // pb == (t-1)&1

    // ---- Layer 0: recurrent matmul + rank-1 input feed -------------------
    float xb8[8];
    #pragma unroll
    for (int r = 0; r < 8; ++r) {
      const int b = m0 + lhalf * 8 + r;
      xb8[r] = (t < kT) ? iput[b * kT + t] : (outAcc[pb * kB + b] + blin0);
    }
    layer_pass<false, false>(hbuf + (size_t)(p * kL + 0) * LBH, hbuf,
                             hbuf + (size_t)(np * kL + 0) * LBH,
                             WhhB + (size_t)0 * kG * kH, WihB,
                             bias[0], xb8, wi0v, creg[0], 0.f, nullptr,
                             mrow, hcol0, lrow, lhalf, hcol, m0);
    prefetch_layer(WhhB + (size_t)1 * kG * kH, WihB + (size_t)0 * kG * kH,
                   true, hcol0, lrow, lhalf);
    grid_sync(bar, epoch);

    // ---- During layer 1: wave 0 drains + re-zeros last step's outAcc -----
    if (wid == 0 && t > 0) {
      #pragma unroll
      for (int b = 0; b < kB; b += 32) {
        const int bb = b + lane;
        out[bb * kSteps + (t - 1)] = outAcc[pb * kB + bb] + blin0;
        outAcc[pb * kB + bb] = 0.f;
      }
    }
    layer_pass<true, false>(hbuf + (size_t)(p * kL + 1) * LBH,
                            hbuf + (size_t)(np * kL + 0) * LBH,
                            hbuf + (size_t)(np * kL + 1) * LBH,
                            WhhB + (size_t)1 * kG * kH,
                            WihB + (size_t)0 * kG * kH,
                            bias[1], xb8, wi0v, creg[1], 0.f, nullptr,
                            mrow, hcol0, lrow, lhalf, hcol, m0);
    prefetch_layer(WhhB + (size_t)2 * kG * kH, WihB + (size_t)1 * kG * kH,
                   true, hcol0, lrow, lhalf);
    grid_sync(bar, epoch);

    // ---- Layer 2 ---------------------------------------------------------
    layer_pass<true, false>(hbuf + (size_t)(p * kL + 2) * LBH,
                            hbuf + (size_t)(np * kL + 1) * LBH,
                            hbuf + (size_t)(np * kL + 2) * LBH,
                            WhhB + (size_t)2 * kG * kH,
                            WihB + (size_t)1 * kG * kH,
                            bias[2], xb8, wi0v, creg[2], 0.f, nullptr,
                            mrow, hcol0, lrow, lhalf, hcol, m0);
    prefetch_layer(WhhB + (size_t)3 * kG * kH, WihB + (size_t)2 * kG * kH,
                   true, hcol0, lrow, lhalf);
    grid_sync(bar, epoch);

    // ---- Layer 3 + fused readout partials into outAcc[t&1] ---------------
    layer_pass<true, true>(hbuf + (size_t)(p * kL + 3) * LBH,
                           hbuf + (size_t)(np * kL + 2) * LBH,
                           hbuf + (size_t)(np * kL + 3) * LBH,
                           WhhB + (size_t)3 * kG * kH,
                           WihB + (size_t)2 * kG * kH,
                           bias[3], xb8, wi0v, creg[3], wlinv,
                           outAcc + (size_t)p * kB,
                           mrow, hcol0, lrow, lhalf, hcol, m0);
    prefetch_layer(WhhB, nullptr, false, hcol0, lrow, lhalf);  // next step L0
    grid_sync(bar, epoch);
  }

  // Final timestep's readout (t = 319 lives in outAcc[1]).
  if (wid == 0) {
    #pragma unroll
    for (int b = 0; b < kB; b += 32) {
      const int bb = b + lane;
      out[bb * kSteps + (kSteps - 1)] = outAcc[1 * kB + bb] + blin0;
    }
  }
}

extern "C" void kernel_launch(void* const* d_in, const int* in_sizes, int n_in,
                              void* d_out, int out_size, void* d_ws, size_t ws_size,
                              hipStream_t stream) {
  (void)in_sizes; (void)n_in; (void)out_size; (void)ws_size;
  const float* iput = (const float*)d_in[0];
  const float* Wi0  = (const float*)d_in[1];
  const float* Wih  = (const float*)d_in[2];
  const float* Whh  = (const float*)d_in[3];
  const float* bih  = (const float*)d_in[4];
  const float* bhh  = (const float*)d_in[5];
  const float* Wlin = (const float*)d_in[6];
  const float* blin = (const float*)d_in[7];
  float* out = (float*)d_out;

  // Workspace layout (bytes):
  //   [0,256)              barrier counter
  //   [256,768)            outAcc f32 [2][64]
  //   [768, +524288)       hbuf bf16 [2][4][64][512]
  //   [525056, +8388608)   WhhB bf16
  //   [8913664, +6291456)  WihB bf16     total ~14.5 MB
  unsigned char* ws = (unsigned char*)d_ws;
  unsigned* bar  = (unsigned*)ws;
  float*   oacc  = (float*)(ws + 256);
  __bf16*  hbuf  = (__bf16*)(ws + 768);
  __bf16*  whhB  = (__bf16*)(ws + 525056);
  __bf16*  wihB  = (__bf16*)(ws + 8913664);

  hipMemsetAsync(ws, 0, 525056, stream);  // bar + outAcc + h state

  cvt_weights_kernel<<<28672, 256, 0, stream>>>(Whh, Wih, whhB, wihB);

  lstm_persistent_kernel<<<kNBlk, kNThr, 0, stream>>>(
      iput, Wi0, bih, bhh, Wlin, blin, whhB, wihB, hbuf, oacc, bar, out);
}
